// DepthAwareEPIBranch_13142599926104
// MI455X (gfx1250) — matrix operand order repaired and checked by reading, not compiled, not run on hardware
//
#include <hip/hip_runtime.h>
#include <math.h>

typedef __attribute__((ext_vector_type(16))) __bf16 v16bf;
typedef __attribute__((ext_vector_type(8)))  float  v8f;
typedef unsigned short u16;
typedef __attribute__((ext_vector_type(8))) u16 u16x8;

#define C_DIM   128
#define H_DIM   320
#define W_DIM   320
#define HW      (H_DIM * W_DIM)          // 102400
#define PX_TILE 128                      // pixels per workgroup
#define NWAVES  8

// LDS row pitches (elements), padded +8 to rotate banks by 4 dwords/row
#define P136 136   // 128-wide bf16 rows
#define P264 264   // 256-wide bf16 rows (combined weights)
#define P40  40    // 32-wide bf16 rows
#define P132 132   // 128-wide f32 rows (epilogue reuse)

// LDS byte offsets (all 16B-aligned)
#define OFF_WC   0                                   // [128][264] u16 = 67584
#define OFF_W1   (OFF_WC  + 128 * P264 * 2)          // [ 32][136] u16 =  8704
#define OFF_W2   (OFF_W1  +  32 * P136 * 2)          // [128][ 40] u16 = 10240
#define OFF_AEH  (OFF_W2  + 128 * P40  * 2)          // [128][136] u16 = 34816
#define OFF_AEV  (OFF_AEH + PX_TILE * P136 * 2)      // [128][136] u16 = 34816
#define OFF_EPI  (OFF_AEV + PX_TILE * P136 * 2)      // [128][136] u16 = 34816
#define OFF_T    (OFF_EPI + PX_TILE * P136 * 2)      // [128][ 40] u16 = 10240
#define SMEM_SZ  (OFF_T   + PX_TILE * P40  * 2)      // 201216 bytes
#define OFF_S    OFF_AEH                             // reuse AEH+AEV: 128*132*4 = 67584 <= 69632

__device__ __forceinline__ u16 f2bf(float f) {
  unsigned u = __float_as_uint(f);
  u += 0x7FFFu + ((u >> 16) & 1u);     // round-to-nearest-even
  return (u16)(u >> 16);
}

union FragBF { u16x8 h[2]; v16bf v; };

// Prologue: fold pointwise conv weights into fuse weights.
// wc[o][i<128]  = sum_j w_fuse[o][j]     * w_h_pw[j][i]
// wc[o][128+i]  = sum_j w_fuse[o][128+j] * w_v_pw[j][i]
__global__ __launch_bounds__(256) void combine_weights(
    const float* __restrict__ w_h_pw, const float* __restrict__ w_v_pw,
    const float* __restrict__ w_fuse, u16* __restrict__ wc) {
  int o = blockIdx.x;       // 0..127
  int i = threadIdx.x;      // 0..255
  float s = 0.f;
  if (i < 128) {
    for (int j = 0; j < 128; ++j) s += w_fuse[o * 256 + j] * w_h_pw[j * 128 + i];
  } else {
    int ii = i - 128;
    for (int j = 0; j < 128; ++j) s += w_fuse[o * 256 + 128 + j] * w_v_pw[j * 128 + ii];
  }
  wc[o * 256 + i] = f2bf(s);
}

__global__ __launch_bounds__(256, 1) void fused_epi_kernel(
    const float* __restrict__ x,
    const float* __restrict__ w_h_dw,   // [128][5]
    const float* __restrict__ w_v_dw,   // [128][5]
    const float* __restrict__ w_dm1,    // [32][128]
    const float* __restrict__ w_dm2,    // [128][32]
    const float* __restrict__ scale,
    const u16*   __restrict__ wc,       // [128][256] bf16 combined
    float* __restrict__ out) {
  extern __shared__ char smem[];
  u16* sWc  = (u16*)(smem + OFF_WC);
  u16* sW1  = (u16*)(smem + OFF_W1);
  u16* sW2  = (u16*)(smem + OFF_W2);
  u16* sAeh = (u16*)(smem + OFF_AEH);
  u16* sAev = (u16*)(smem + OFF_AEV);
  u16* sEpi = (u16*)(smem + OFF_EPI);
  u16* sT   = (u16*)(smem + OFF_T);
  float* sS = (float*)(smem + OFF_S);

  const int tid   = threadIdx.x;
  const int wv    = tid >> 5;          // wave id 0..7 -> M-tile
  const int lane  = tid & 31;
  const int lhalf = lane & 15;
  const int hi    = (lane >> 4) & 1;
  const int pbase = blockIdx.x * PX_TILE;

  // ---- Stage weights into LDS ----
  for (int idx = tid; idx < 128 * 256; idx += 256) {
    int o = idx >> 8, i = idx & 255;
    sWc[o * P264 + i] = wc[idx];
  }
  for (int idx = tid; idx < 32 * 128; idx += 256) {
    int o = idx >> 7, i = idx & 127;
    sW1[o * P136 + i] = f2bf(w_dm1[idx]);
  }
  for (int idx = tid; idx < 128 * 32; idx += 256) {
    int o = idx >> 5, i = idx & 31;
    sW2[o * P40 + i] = f2bf(w_dm2[idx]);
  }

  // ---- Stage 1: depthwise 1x5 / 5x1 stencils (confined to 5-groups) + lrelu, bf16 to LDS ----
  for (int idx = tid; idx < PX_TILE * C_DIM; idx += 256) {
    int p  = idx & (PX_TILE - 1);
    int c  = idx >> 7;
    int pg = pbase + p;
    int b  = (pg >= HW) ? 1 : 0;
    int r  = pg - b * HW;
    int y  = r / W_DIM;
    int xc = r - y * W_DIM;
    const float* chan = x + (size_t)(b * C_DIM + c) * HW;
    const float* row  = chan + y * W_DIM;

    int lox = (xc / 5) * 5, px5 = xc - lox;
    float sh = 0.f;
#pragma unroll
    for (int t = 0; t < 5; ++t) {
      int q = px5 - 2 + t;
      if (q >= 0 && q < 5) sh += w_h_dw[c * 5 + t] * row[lox + q];
    }
    sh = sh > 0.f ? sh : 0.1f * sh;
    sAeh[p * P136 + c] = f2bf(sh);

    int loy = (y / 5) * 5, py5 = y - loy;
    float sv = 0.f;
#pragma unroll
    for (int t = 0; t < 5; ++t) {
      int q = py5 - 2 + t;
      if (q >= 0 && q < 5) sv += w_v_dw[c * 5 + t] * chan[(loy + q) * W_DIM + xc];
    }
    sv = sv > 0.f ? sv : 0.1f * sv;
    sAev[p * P136 + c] = f2bf(sv);
  }
  __syncthreads();

  // ---- GEMM1: epi = [A_eh | A_ev](128x256) x Wc^T -> 128x128, K in 8 chunks of 32 ----
  v8f acc[8];
#pragma unroll
  for (int i = 0; i < 8; ++i)
#pragma unroll
    for (int j = 0; j < 8; ++j) acc[i][j] = 0.f;

  const int arow = 16 * wv + lhalf;
#pragma unroll
  for (int kc = 0; kc < 8; ++kc) {
    const u16* abase = (kc < 4) ? sAeh : sAev;
    int kb = (kc < 4) ? 32 * kc : 32 * kc - 128;
    FragBF fa;
    fa.h[0] = *(const u16x8*)(abase + arow * P136 + kb + hi * 8);
    fa.h[1] = *(const u16x8*)(abase + arow * P136 + kb + 16 + hi * 8);
    int kcol = 32 * kc + hi * 8;
#pragma unroll
    for (int nt = 0; nt < 8; ++nt) {
      int o = 16 * nt + lhalf;
      FragBF fb;
      fb.h[0] = *(const u16x8*)(sWc + o * P264 + kcol);
      fb.h[1] = *(const u16x8*)(sWc + o * P264 + kcol + 16);
      acc[nt] = __builtin_amdgcn_wmma_f32_16x16x32_bf16(
          false, fa.v, false, fb.v, (short)0, acc[nt], false, false);
    }
  }

  // epi -> LDS bf16 (A operand for dm1); keep fp32 epi in registers
#pragma unroll
  for (int nt = 0; nt < 8; ++nt) {
    int n = 16 * nt + lhalf;
#pragma unroll
    for (int j = 0; j < 8; ++j) {
      int m = 16 * wv + j + hi * 8;
      sEpi[m * P136 + n] = f2bf(acc[nt][j]);
    }
  }
  __syncthreads();

  // ---- GEMM2: t = lrelu(epi x W1^T), N=32 ----
  v8f acc2[2];
#pragma unroll
  for (int i = 0; i < 2; ++i)
#pragma unroll
    for (int j = 0; j < 8; ++j) acc2[i][j] = 0.f;

#pragma unroll
  for (int kc = 0; kc < 4; ++kc) {
    int kb = 32 * kc;
    FragBF fa;
    fa.h[0] = *(const u16x8*)(sEpi + arow * P136 + kb + hi * 8);
    fa.h[1] = *(const u16x8*)(sEpi + arow * P136 + kb + 16 + hi * 8);
    int kcol = kb + hi * 8;
#pragma unroll
    for (int nt = 0; nt < 2; ++nt) {
      int o = 16 * nt + lhalf;
      FragBF fb;
      fb.h[0] = *(const u16x8*)(sW1 + o * P136 + kcol);
      fb.h[1] = *(const u16x8*)(sW1 + o * P136 + kcol + 16);
      acc2[nt] = __builtin_amdgcn_wmma_f32_16x16x32_bf16(
          false, fa.v, false, fb.v, (short)0, acc2[nt], false, false);
    }
  }
#pragma unroll
  for (int nt = 0; nt < 2; ++nt) {
    int n = 16 * nt + lhalf;
#pragma unroll
    for (int j = 0; j < 8; ++j) {
      int m = 16 * wv + j + hi * 8;
      float z = acc2[nt][j];
      z = z > 0.f ? z : 0.1f * z;
      sT[m * P40 + n] = f2bf(z);
    }
  }
  __syncthreads();

  // ---- GEMM3: d = t x W2^T (K=32), sigmoid, multiply with live epi, scale ----
  v8f acc3[8];
#pragma unroll
  for (int i = 0; i < 8; ++i)
#pragma unroll
    for (int j = 0; j < 8; ++j) acc3[i][j] = 0.f;

  {
    FragBF fa;
    fa.h[0] = *(const u16x8*)(sT + arow * P40 + hi * 8);
    fa.h[1] = *(const u16x8*)(sT + arow * P40 + 16 + hi * 8);
#pragma unroll
    for (int nt = 0; nt < 8; ++nt) {
      int o = 16 * nt + lhalf;
      FragBF fb;
      fb.h[0] = *(const u16x8*)(sW2 + o * P40 + hi * 8);
      fb.h[1] = *(const u16x8*)(sW2 + o * P40 + 16 + hi * 8);
      acc3[nt] = __builtin_amdgcn_wmma_f32_16x16x32_bf16(
          false, fa.v, false, fb.v, (short)0, acc3[nt], false, false);
    }
  }

  const float sc = scale[0];
#pragma unroll
  for (int nt = 0; nt < 8; ++nt) {
    int n = 16 * nt + lhalf;
#pragma unroll
    for (int j = 0; j < 8; ++j) {
      int m = 16 * wv + j + hi * 8;
      float dw = 1.f / (1.f + __expf(-acc3[nt][j]));
      sS[m * P132 + n] = sc * acc[nt][j] * dw;   // scale * epi * depth_w
    }
  }
  __syncthreads();

  // ---- Epilogue: out = x + s (coalesced along pixels; x is L2-resident) ----
  for (int idx = tid; idx < PX_TILE * C_DIM; idx += 256) {
    int p  = idx & (PX_TILE - 1);
    int c  = idx >> 7;
    int pg = pbase + p;
    int b  = (pg >= HW) ? 1 : 0;
    int r  = pg - b * HW;
    size_t addr = (size_t)(b * C_DIM + c) * HW + r;
    out[addr] = x[addr] + sS[p * P132 + c];
  }
}

extern "C" void kernel_launch(void* const* d_in, const int* in_sizes, int n_in,
                              void* d_out, int out_size, void* d_ws, size_t ws_size,
                              hipStream_t stream) {
  const float* x      = (const float*)d_in[0];
  const float* w_h_dw = (const float*)d_in[1];
  const float* w_h_pw = (const float*)d_in[2];
  const float* w_v_dw = (const float*)d_in[3];
  const float* w_v_pw = (const float*)d_in[4];
  const float* w_dm1  = (const float*)d_in[5];
  const float* w_dm2  = (const float*)d_in[6];
  const float* w_fuse = (const float*)d_in[7];
  const float* scale  = (const float*)d_in[8];
  float* out = (float*)d_out;
  u16*   wc  = (u16*)d_ws;   // 128*256 bf16 = 64 KB of scratch

  combine_weights<<<128, 256, 0, stream>>>(w_h_pw, w_v_pw, w_fuse, wc);

  const int total_px = 2 * HW;                 // 204800
  const int grid = total_px / PX_TILE;         // 1600, exact
  fused_epi_kernel<<<grid, 256, SMEM_SZ, stream>>>(
      x, w_h_dw, w_v_dw, w_dm1, w_dm2, scale, wc, out);
}
